// Simple_Patch_44942537785435
// MI455X (gfx1250) — matrix operand order repaired and checked by reading, not compiled, not run on hardware
//
#include <hip/hip_runtime.h>
#include <hip/hip_bf16.h>
#include <math.h>

typedef __attribute__((ext_vector_type(16))) _Float16 v16h;
typedef __attribute__((ext_vector_type(8)))  float    v8f;

#define IMG    224
#define FEAT   112          // IMG / 2 (conv stride 2)
#define NTOK   (FEAT*FEAT)  // 12544 tokens per image
#define EMBED  192
#define NTILE  (EMBED/16)   // 12 WMMA N-tiles
#define KDIM   27           // 3 ch * 3 * 3 (padded to 32 for WMMA K)
#define BATCH  32
#define OUTI   448          // 112 patches * 4 samples
#define WAVES  4            // waves (16-token groups) per block

// GELU via the gfx1250 hardware tanh transcendental (branch-free, ~5 ops vs
// the ~40-op divergent erff expansion).  v_nop after the TRANS op per the
// CDNA5 multicycle co-execution hazard rule (compiler can't see inside asm).
__device__ __forceinline__ float gelu_fast(float x) {
  float t = 0.7978845608028654f * (x + 0.044715f * x * x * x);
  float r;
  asm volatile("v_tanh_f32 %0, %1\n\tv_nop" : "=v"(r) : "v"(t));
  return 0.5f * x * (1.0f + r);
}

// ---------------------------------------------------------------------------
// Pre-pack conv weights [192][27] f32 -> f16 B fragments in a contiguous-K per-lane
// register image for V_WMMA_F32_16X16X32_F16 (B is 32x16, K padded 27->32).
// Layout: lanes 0-15 hold N=lane, K=0..15 (vgpr v -> K=2v,2v+1);
//         lanes 16-31 hold N=lane-16, K=16..31.
// ws image: [tile][lane][8 dwords]  (12*32*8 dwords = 12 KiB, L2-resident)
// ---------------------------------------------------------------------------
__global__ void dp_pack_weights(const float* __restrict__ conv_w,
                                unsigned int* __restrict__ wsB) {
  int tile = blockIdx.x;          // 0..11
  int lane = threadIdx.x;         // 0..31
  int l16 = lane & 15, h = lane >> 4;
  int n = tile * 16 + l16;        // embed channel
  unsigned int d[8];
  #pragma unroll
  for (int v = 0; v < 8; ++v) {
    int k0 = h * 16 + 2 * v;
    _Float16 e0 = (k0     < KDIM) ? (_Float16)conv_w[n * KDIM + k0]     : (_Float16)0.0f;
    _Float16 e1 = (k0 + 1 < KDIM) ? (_Float16)conv_w[n * KDIM + k0 + 1] : (_Float16)0.0f;
    unsigned short u0, u1;
    __builtin_memcpy(&u0, &e0, 2);
    __builtin_memcpy(&u1, &e1, 2);
    d[v] = (unsigned int)u0 | ((unsigned int)u1 << 16);
  }
  unsigned int* dst = wsB + (tile * 32 + lane) * 8;
  *(uint4*)(dst)     = make_uint4(d[0], d[1], d[2], d[3]);
  *(uint4*)(dst + 4) = make_uint4(d[4], d[5], d[6], d[7]);
}

// ---------------------------------------------------------------------------
// Fused: conv3x3 s2 p1 (implicit GEMM via WMMA) -> +bias -> GELU ->
// 192->2 linear -> *PATCH_SIZE.  One wave handles 16 tokens (one feature-row
// strip); 12 WMMAs (fully unrolled) cover EMBED=192.  Only the 2-float
// offset/token hits HBM (3.2 MB vs 308 MB for a materialized token tensor).
// ---------------------------------------------------------------------------
__global__ __launch_bounds__(32 * WAVES) void dp_offsets(
    const float* __restrict__ x,          // [B,3,224,224]
    const unsigned int* __restrict__ wsB, // packed B fragments
    const float* __restrict__ conv_b,     // [192]
    const float* __restrict__ off_w,      // [2,192]
    float* __restrict__ off_out) {        // [B, NTOK, 2]
  __shared__ float stage[WAVES][3][3][36];   // 3ch x 3rows x 34cols (+pad)
  __shared__ float red[WAVES][32][16];

  int lane = threadIdx.x & 31;
  int wv   = threadIdx.x >> 5;
  int group = blockIdx.x * WAVES + wv;       // 16-token group id
  int tokBase = group * 16;
  int b  = tokBase / NTOK;
  int t0 = tokBase % NTOK;
  int fy  = t0 / FEAT;                       // same for all 16 tokens
  int fx0 = t0 % FEAT;                       // multiple of 16
  int l16 = lane & 15, h = lane >> 4;

  // ---- stage the shared input strip into LDS (border-zero = conv padding)
  int iy0 = 2 * fy - 1, ix0 = 2 * fx0 - 1;
  const float* xb = x + (size_t)b * 3 * IMG * IMG;
  for (int i = lane; i < 3 * 3 * 34; i += 32) {
    int c  = i / (3 * 34);
    int r  = i % (3 * 34);
    int ky = r / 34, j = r % 34;
    int gy = iy0 + ky, gx = ix0 + j;
    float v = 0.0f;
    if (gy >= 0 && gy < IMG && gx >= 0 && gx < IMG)
      v = xb[((size_t)c * IMG + gy) * IMG + gx];
    stage[wv][c][ky][j] = v;
  }
  __syncthreads();

  // ---- build A fragment: 16 tokens x K32 f16 (contiguous K = 16*h + s, matching packed B)
  v16h a;
  #pragma unroll
  for (int v = 0; v < 8; ++v) {
    int kb = h * 16 + 2 * v;
    #pragma unroll
    for (int e = 0; e < 2; ++e) {
      int k = kb + e;
      float val = 0.0f;
      if (k < KDIM) {
        int c  = k / 9;
        int r  = k % 9;
        int ky = r / 3, kx = r % 3;
        val = stage[wv][c][ky][2 * l16 + kx];
      }
      a[2 * v + e] = (_Float16)val;
    }
  }

  float o0[8], o1[8];
  #pragma unroll
  for (int r = 0; r < 8; ++r) { o0[r] = 0.0f; o1[r] = 0.0f; }

  // ---- 12 WMMAs; consume each 16x16 C tile immediately (bias+GELU+linear).
  // Fully unrolled: B-fragment b128 loads clause up, and the epilogue VALU of
  // tile t co-executes with the XDL pipe running tile t+1's WMMA.
  #pragma unroll
  for (int tile = 0; tile < NTILE; ++tile) {
    const uint4* bp = (const uint4*)(wsB + ((size_t)tile * 32 + lane) * 8);
    union { uint4 u[2]; v16h hv; } bm;
    bm.u[0] = bp[0];
    bm.u[1] = bp[1];
    v8f cz = {};
    v8f acc = __builtin_amdgcn_wmma_f32_16x16x32_f16(
        false, a, false, bm.hv, (short)0, cz, false, false);

    int n = tile * 16 + l16;               // embed channel (C tile N = l16)
    float w0   = off_w[n];                 // off_w[0][n] -> dx
    float w1   = off_w[EMBED + n];         // off_w[1][n] -> dy
    float bias = conv_b[n];
    #pragma unroll
    for (int r = 0; r < 8; ++r) {          // C tile M = r + 8*h
      float g = gelu_fast(acc[r] + bias);
      o0[r] += g * w0;
      o1[r] += g * w1;
    }
  }

  // ---- reduce across the 16 lanes of each half (distinct embed channels)
  #pragma unroll
  for (int r = 0; r < 8; ++r) {
    red[wv][lane][2 * r + 0] = o0[r];
    red[wv][lane][2 * r + 1] = o1[r];
  }
  __syncthreads();
  {
    int tr = lane & 15;        // token row within group
    int o  = lane >> 4;        // 0 = dx, 1 = dy
    int hh = tr >> 3, r = tr & 7;
    float s = 0.0f;
    #pragma unroll
    for (int t = 0; t < 16; ++t) s += red[wv][hh * 16 + t][2 * r + o];
    // * PATCH_SIZE (pixel-space offsets)
    off_out[((size_t)b * NTOK + t0 + tr) * 2 + o] = s * 2.0f;
  }
}

// ---------------------------------------------------------------------------
// Deformable bilinear resampling.  One thread = one 4-pixel patch-row of the
// output (offset fetched once, float4 store, 16B aligned: row = 448 floats).
// ---------------------------------------------------------------------------
__global__ __launch_bounds__(256) void dp_resample(
    const float* __restrict__ x,    // [B,3,224,224]
    const float* __restrict__ off,  // [B,NTOK,2]
    float* __restrict__ out) {      // [B,3,448,448]
  long idx = (long)blockIdx.x * blockDim.x + threadIdx.x;
  const long total = (long)BATCH * 3 * OUTI * FEAT;
  if (idx >= total) return;

  int  pw = (int)(idx % FEAT);
  long t  = idx / FEAT;
  int  oy = (int)(t % OUTI);  t /= OUTI;
  int  c  = (int)(t % 3);     t /= 3;
  int  b  = (int)t;
  int  ph = oy >> 2, py = oy & 3;

  const float* ob = off + ((size_t)b * NTOK + (size_t)ph * FEAT + pw) * 2;
  float dx = ob[0], dy = ob[1];

  float cy = (ph + 0.5f) * 2.0f;
  float cx = (pw + 0.5f) * 2.0f;
  float ys = cy + dy + ((py + 0.5f) * 0.5f - 1.0f);

  const float* img = x + ((size_t)b * 3 + c) * IMG * IMG;
  float y0f = floorf(ys);
  float wy  = ys - y0f;
  int y0 = (int)y0f;  y0 = min(max(y0, 0), IMG - 1);
  int y1 = min(y0 + 1, IMG - 1);
  const float* row0 = img + (size_t)y0 * IMG;
  const float* row1 = img + (size_t)y1 * IMG;

  float4 res;
  float* rp = (float*)&res;
  #pragma unroll
  for (int px = 0; px < 4; ++px) {
    float xs  = cx + dx + ((px + 0.5f) * 0.5f - 1.0f);
    float x0f = floorf(xs);
    float wx  = xs - x0f;
    int x0 = (int)x0f;  x0 = min(max(x0, 0), IMG - 1);
    int x1 = min(x0 + 1, IMG - 1);
    float v00 = row0[x0], v01 = row0[x1];
    float v10 = row1[x0], v11 = row1[x1];
    rp[px] = v00 * (1.0f - wy) * (1.0f - wx) + v01 * (1.0f - wy) * wx
           + v10 * wy * (1.0f - wx)          + v11 * wy * wx;
  }
  float4* op = (float4*)(out + (((size_t)(b * 3 + c) * OUTI + oy) * OUTI) + (size_t)pw * 4);
  *op = res;
}

// ---------------------------------------------------------------------------
extern "C" void kernel_launch(void* const* d_in, const int* in_sizes, int n_in,
                              void* d_out, int out_size, void* d_ws, size_t ws_size,
                              hipStream_t stream) {
  const float* x      = (const float*)d_in[0];  // [32,3,224,224]
  const float* conv_w = (const float*)d_in[1];  // [192,3,3,3]
  const float* conv_b = (const float*)d_in[2];  // [192]
  const float* off_w  = (const float*)d_in[3];  // [2,192]
  float* out = (float*)d_out;                   // [32,3,448,448]

  unsigned int* wsB = (unsigned int*)d_ws;                   // 12 KiB packed B
  float* off = (float*)((char*)d_ws + NTILE * 32 * 8 * 4);   // [B,NTOK,2] ~3.2 MB

  // 1) pack conv weights into WMMA B-fragment image (deterministic, cheap)
  dp_pack_weights<<<NTILE, 32, 0, stream>>>(conv_w, wsB);

  // 2) fused conv -> GELU -> 192->2 linear -> per-patch offsets
  int groups = BATCH * (NTOK / 16);                 // 25088 16-token groups
  dp_offsets<<<groups / WAVES, 32 * WAVES, 0, stream>>>(x, wsB, conv_b, off_w, off);

  // 3) deformable bilinear resampling
  long total = (long)BATCH * 3 * OUTI * FEAT;
  dp_resample<<<(int)((total + 255) / 256), 256, 0, stream>>>(x, off, out);
}
